// GraphFusion_58643483460165
// MI455X (gfx1250) — compile-verified, matrix operand order checked by reading, not run on hardware
//
#include <hip/hip_runtime.h>
#include <hip/hip_bf16.h>

typedef __attribute__((ext_vector_type(16))) __bf16 v16bf;
typedef __attribute__((ext_vector_type(8)))  __bf16 v8bf;
typedef __attribute__((ext_vector_type(8)))  float  v8f;

#define BATCH   4
#define HEADS   8
#define DHEAD   64
#define NTOK    12281
#define MPATH   281
#define NHIST   12000          // NTOK - MPATH
#define DIM     512
#define NQKV    1536
#define MROWS   (BATCH*NTOK)   // 49124
#define VSTRIDE 12312          // padded token stride for V^T: mult of 8, >= 280+376*32

// ---------------------------------------------------------------- helpers ---

__device__ __forceinline__ v8f zero8() {
  v8f z;
  #pragma unroll
  for (int i = 0; i < 8; ++i) z[i] = 0.0f;
  return z;
}

// Load one 16x32 bf16 WMMA fragment (A- or B-side) for this lane.
// Caller passes p = row_base + k0 + bk  where bk = (lane>>4)*8, row = lane&15.
// VGPR0..3 <- 8 contiguous bf16 at p (K = bk..bk+7), VGPR4..7 <- at p+16.
__device__ __forceinline__ v16bf ld_frag(const __bf16* p) {
  v8bf lo = *(const v8bf*)(p);
  v8bf hi = *(const v8bf*)(p + 16);
  v16bf f;
  #pragma unroll
  for (int i = 0; i < 8; ++i) { f[i] = lo[i]; f[i + 8] = hi[i]; }
  return f;
}

__device__ __forceinline__ v8f wmma_bf16(v16bf a, v16bf b, v8f c) {
  // (neg_a, A, neg_b, B, c_mod, C, reuse_a, reuse_b)
  return __builtin_amdgcn_wmma_f32_16x16x32_bf16(false, a, false, b, (short)0, c, false, false);
}

// Reductions across the 16 lanes holding one C/D-matrix row (xor masks stay
// inside each 16-lane half of the wave32).
__device__ __forceinline__ float rowmax16(float v) {
  v = fmaxf(v, __shfl_xor(v, 1, 32));
  v = fmaxf(v, __shfl_xor(v, 2, 32));
  v = fmaxf(v, __shfl_xor(v, 4, 32));
  v = fmaxf(v, __shfl_xor(v, 8, 32));
  return v;
}
__device__ __forceinline__ float rowsum16(float v) {
  v += __shfl_xor(v, 1, 32);
  v += __shfl_xor(v, 2, 32);
  v += __shfl_xor(v, 4, 32);
  v += __shfl_xor(v, 8, 32);
  return v;
}

// ---------------------------------------------------------- K1: conversions

__global__ void cvt_x_kernel(const float* __restrict__ x, __bf16* __restrict__ xb, int n) {
  int i = blockIdx.x * blockDim.x + threadIdx.x;
  if (i < n) xb[i] = (__bf16)x[i];
}

// WT[n][k] = W[k][n], bf16 (so GEMM B-fragments are K-contiguous)
__global__ void transpose_w_kernel(const float* __restrict__ W, __bf16* __restrict__ WT) {
  int i = blockIdx.x * blockDim.x + threadIdx.x;
  if (i < NQKV * DIM) {
    int n = i >> 9;        // / DIM
    int k = i & (DIM - 1); // % DIM
    WT[i] = (__bf16)W[(size_t)k * NQKV + n];
  }
}

// -------------------------------------------------- K2: QKV GEMM (bf16 WMMA)
// Each wave computes a 32x64 tile of qkv = X(49124x512) @ W(512x1536) with a
// software-pipelined (double-buffered) K loop so loads overlap WMMAs.
// Writes split outputs: Q (scaled) / K as [bh][tok][d], V transposed [bh][d][tok].

__global__ __launch_bounds__(128) void qkv_gemm_kernel(
    const __bf16* __restrict__ XB, const __bf16* __restrict__ WT,
    __bf16* __restrict__ Qb, __bf16* __restrict__ Kb, __bf16* __restrict__ VTb) {
  const int MT  = (MROWS + 31) / 32;  // 1536
  const int NTW = NQKV / 64;          // 24
  int lane = threadIdx.x & 31;
  int wid  = blockIdx.x * (blockDim.x >> 5) + (threadIdx.x >> 5);
  if (wid >= MT * NTW) return;
  int mt = wid / NTW, ntw = wid % NTW;
  int row0 = mt * 32, n0 = ntw * 64;
  int r = lane & 15, lh = lane >> 4, bk = lh * 8;

  int arow0 = row0 + r, arow1 = row0 + 16 + r;
  const __bf16* aptr0 = XB + (size_t)(arow0 < MROWS ? arow0 : 0) * DIM;
  const __bf16* aptr1 = XB + (size_t)(arow1 < MROWS ? arow1 : 0) * DIM;
  const __bf16* bptr  = WT + (size_t)(n0 + r) * DIM;  // +ct*16*DIM per col tile

  v8f acc[2][4];
  #pragma unroll
  for (int mb = 0; mb < 2; ++mb)
    #pragma unroll
    for (int ct = 0; ct < 4; ++ct) acc[mb][ct] = zero8();

  // prologue: fragments for k0 = 0
  v16bf a0c = ld_frag(aptr0 + bk);
  v16bf a1c = ld_frag(aptr1 + bk);
  v16bf bc[4];
  #pragma unroll
  for (int ct = 0; ct < 4; ++ct) bc[ct] = ld_frag(bptr + (size_t)ct * 16 * DIM + bk);

  // pipelined steps: load k0, consume k0-32
  #pragma unroll
  for (int k0 = 32; k0 < DIM; k0 += 32) {
    v16bf a0n = ld_frag(aptr0 + k0 + bk);
    v16bf a1n = ld_frag(aptr1 + k0 + bk);
    v16bf bn[4];
    #pragma unroll
    for (int ct = 0; ct < 4; ++ct) bn[ct] = ld_frag(bptr + (size_t)ct * 16 * DIM + k0 + bk);

    #pragma unroll
    for (int ct = 0; ct < 4; ++ct) {
      acc[0][ct] = wmma_bf16(a0c, bc[ct], acc[0][ct]);
      acc[1][ct] = wmma_bf16(a1c, bc[ct], acc[1][ct]);
    }
    a0c = a0n; a1c = a1n;
    #pragma unroll
    for (int ct = 0; ct < 4; ++ct) bc[ct] = bn[ct];
  }
  // epilogue: consume final fragments
  #pragma unroll
  for (int ct = 0; ct < 4; ++ct) {
    acc[0][ct] = wmma_bf16(a0c, bc[ct], acc[0][ct]);
    acc[1][ct] = wmma_bf16(a1c, bc[ct], acc[1][ct]);
  }

  // C/D layout: element (vgpr rr, lane) -> row = row0+mb*16+rr+8*lh,
  //                                       col = n0+ct*16+(lane&15)
  #pragma unroll
  for (int mb = 0; mb < 2; ++mb) {
    #pragma unroll
    for (int rr = 0; rr < 8; ++rr) {
      int m = row0 + mb * 16 + rr + 8 * lh;
      if (m >= MROWS) continue;
      int b = m / NTOK, tok = m % NTOK;
      #pragma unroll
      for (int ct = 0; ct < 4; ++ct) {
        int c = n0 + ct * 16 + r;
        float val = acc[mb][ct][rr];
        if (c < 512) {                    // Q, fold softmax scale d^-0.5 = 0.125
          int h = c >> 6, d = c & 63;
          Qb[((size_t)(b * HEADS + h) * NTOK + tok) * DHEAD + d] = (__bf16)(val * 0.125f);
        } else if (c < 1024) {            // K
          int cc = c - 512; int h = cc >> 6, d = cc & 63;
          Kb[((size_t)(b * HEADS + h) * NTOK + tok) * DHEAD + d] = (__bf16)val;
        } else {                          // V, transposed [bh][d][tok]
          int cc = c - 1024; int h = cc >> 6, d = cc & 63;
          VTb[((size_t)(b * HEADS + h) * DHEAD + d) * VSTRIDE + tok] = (__bf16)val;
        }
      }
    }
  }
}

// ------------------------- K3: attn_hist = softmax(q_hist @ k_path^T) @ v_path
// One wave handles 16 hist rows vs all 281 path cols (padded to 288).
// Output goes to tokens [0, 12000) of d_out.

__global__ __launch_bounds__(128) void attn_hist_kernel(
    const __bf16* __restrict__ Qb, const __bf16* __restrict__ Kb,
    const __bf16* __restrict__ VTb, float* __restrict__ out) {
  __shared__ __align__(16) __bf16 lds[4][16 * 288];
  const int RT = NHIST / 16;  // 750
  int lane = threadIdx.x & 31, wslot = threadIdx.x >> 5;
  int wid = blockIdx.x * 4 + wslot;
  if (wid >= BATCH * HEADS * RT) return;
  int bh = wid / RT, rt = wid % RT;
  int row0 = MPATH + rt * 16;  // hist rows live at tokens [281, 12281)
  int r = lane & 15, lh = lane >> 4, bk = lh * 8;

  const __bf16* Qh = Qb  + (size_t)bh * NTOK * DHEAD;
  const __bf16* Kh = Kb  + (size_t)bh * NTOK * DHEAD;
  const __bf16* Vh = VTb + (size_t)bh * DHEAD * VSTRIDE;

  v16bf aq0 = ld_frag(Qh + (size_t)(row0 + r) * DHEAD + bk);        // K = 0..31
  v16bf aq1 = ld_frag(Qh + (size_t)(row0 + r) * DHEAD + 32 + bk);   // K = 32..63

  // ---- scores: 18 tiles of 16 cols, kept in registers
  v8f sc[18];
  #pragma unroll
  for (int ct = 0; ct < 18; ++ct) {
    int col = ct * 16 + r;  // B column = path token (tokens 281..287 masked below)
    v16bf b0 = ld_frag(Kh + (size_t)col * DHEAD + bk);
    v16bf b1 = ld_frag(Kh + (size_t)col * DHEAD + 32 + bk);
    v8f s = zero8();
    s = wmma_bf16(aq0, b0, s);
    s = wmma_bf16(aq1, b1, s);
    sc[ct] = s;
  }

  // ---- masked row softmax over 281 cols
  float rm[8], rs[8];
  #pragma unroll
  for (int rr = 0; rr < 8; ++rr) rm[rr] = -3.0e38f;
  #pragma unroll
  for (int ct = 0; ct < 18; ++ct) {
    bool colok = (ct * 16 + r) < MPATH;
    #pragma unroll
    for (int rr = 0; rr < 8; ++rr) {
      float v = colok ? sc[ct][rr] : -3.0e38f;
      sc[ct][rr] = v;
      rm[rr] = fmaxf(rm[rr], v);
    }
  }
  #pragma unroll
  for (int rr = 0; rr < 8; ++rr) { rm[rr] = rowmax16(rm[rr]); rs[rr] = 0.0f; }
  #pragma unroll
  for (int ct = 0; ct < 18; ++ct) {
    #pragma unroll
    for (int rr = 0; rr < 8; ++rr) {
      float p = __expf(sc[ct][rr] - rm[rr]);
      sc[ct][rr] = p;
      rs[rr] += p;
    }
  }
  #pragma unroll
  for (int rr = 0; rr < 8; ++rr) rs[rr] = 1.0f / rowsum16(rs[rr]);

  // ---- probs -> LDS (row-major 16x288 bf16), then re-read as A fragments
  __bf16* pl = &lds[wslot][0];
  #pragma unroll
  for (int ct = 0; ct < 18; ++ct) {
    #pragma unroll
    for (int rr = 0; rr < 8; ++rr) {
      pl[(rr + 8 * lh) * 288 + ct * 16 + r] = (__bf16)(sc[ct][rr] * rs[rr]);
    }
  }
  asm volatile("s_wait_dscnt 0" ::: "memory");

  // ---- P(16x288) @ V(288x64): V^T rows are d-columns -> contiguous B frags
  v8f acc[4];
  #pragma unroll
  for (int nt = 0; nt < 4; ++nt) acc[nt] = zero8();
  #pragma unroll
  for (int kt = 0; kt < 9; ++kt) {
    int k0 = kt * 32;
    v16bf ap = ld_frag(pl + r * 288 + k0 + bk);
    #pragma unroll
    for (int nt = 0; nt < 4; ++nt) {
      v16bf bv = ld_frag(Vh + (size_t)(nt * 16 + r) * VSTRIDE + k0 + bk);
      acc[nt] = wmma_bf16(ap, bv, acc[nt]);
    }
  }

  // ---- store: hist outputs occupy tokens [0, 12000)
  int b = bh >> 3, h = bh & 7;
  float* ob = out + (size_t)b * NTOK * 512 + (size_t)h * 64;
  #pragma unroll
  for (int rr = 0; rr < 8; ++rr) {
    int tok = (row0 - MPATH) + rr + 8 * lh;  // 0..11999
    #pragma unroll
    for (int nt = 0; nt < 4; ++nt) {
      ob[(size_t)tok * 512 + nt * 16 + r] = acc[nt][rr];
    }
  }
}

// --------------- K4: attn_path = softmax(q_path @ k_hist^T) @ v_hist (flash)
// One wave handles 16 path rows, streaming 12000 hist cols in 32-token chunks
// with online softmax. Chunks start at token 280 (8-aligned for b128 loads);
// token 280 and tokens >= 12281 are masked. Output -> tokens [12000, 12281).

__global__ __launch_bounds__(128) void attn_path_kernel(
    const __bf16* __restrict__ Qb, const __bf16* __restrict__ Kb,
    const __bf16* __restrict__ VTb, float* __restrict__ out) {
  __shared__ __align__(16) __bf16 lds[4][16 * 32];
  const int RT = (MPATH + 15) / 16;  // 18
  int lane = threadIdx.x & 31, wslot = threadIdx.x >> 5;
  int wid = blockIdx.x * 4 + wslot;
  if (wid >= BATCH * HEADS * RT) return;
  int bh = wid / RT, rt = wid % RT;
  int row0 = rt * 16;  // path rows at tokens [0, 281); rows 281..287 discarded
  int r = lane & 15, lh = lane >> 4, bk = lh * 8;

  const __bf16* Qh = Qb  + (size_t)bh * NTOK * DHEAD;
  const __bf16* Kh = Kb  + (size_t)bh * NTOK * DHEAD;
  const __bf16* Vh = VTb + (size_t)bh * DHEAD * VSTRIDE;

  v16bf aq0 = ld_frag(Qh + (size_t)(row0 + r) * DHEAD + bk);
  v16bf aq1 = ld_frag(Qh + (size_t)(row0 + r) * DHEAD + 32 + bk);

  float m[8], l[8];
  v8f acc[4];
  #pragma unroll
  for (int rr = 0; rr < 8; ++rr) { m[rr] = -3.0e38f; l[rr] = 0.0f; }
  #pragma unroll
  for (int nt = 0; nt < 4; ++nt) acc[nt] = zero8();

  __bf16* pl = &lds[wslot][0];
  const int J0 = 280, NCH = 376;  // covers hist tokens [281, 12281)

  for (int jt = 0; jt < NCH; ++jt) {
    int j0 = J0 + jt * 32;

    // prefetch next chunk of K / V^T (global_prefetch_b8)
    if (jt + 1 < NCH) {
      __builtin_prefetch(Kh + (size_t)(j0 + 32 + r + 16 * lh) * DHEAD, 0, 1);
      __builtin_prefetch(Vh + (size_t)(r + 16 * lh) * VSTRIDE + j0 + 32, 0, 1);
    }

    // scores for 32 new columns (two 16-col groups)
    v8f s[2];
    #pragma unroll
    for (int g = 0; g < 2; ++g) {
      int col = j0 + g * 16 + r;
      v16bf b0 = ld_frag(Kh + (size_t)col * DHEAD + bk);
      v16bf b1 = ld_frag(Kh + (size_t)col * DHEAD + 32 + bk);
      v8f sg = zero8();
      sg = wmma_bf16(aq0, b0, sg);
      sg = wmma_bf16(aq1, b1, sg);
      bool ok = (col >= MPATH) && (col < NTOK);
      #pragma unroll
      for (int rr = 0; rr < 8; ++rr) if (!ok) sg[rr] = -3.0e38f;
      s[g] = sg;
    }

    // online softmax update (per C/D-row)
    #pragma unroll
    for (int rr = 0; rr < 8; ++rr) {
      float nm = rowmax16(fmaxf(s[0][rr], s[1][rr]));
      nm = fmaxf(m[rr], nm);
      float rsc = __expf(m[rr] - nm);
      m[rr] = nm;
      float p0 = __expf(s[0][rr] - nm);
      float p1 = __expf(s[1][rr] - nm);
      s[0][rr] = p0; s[1][rr] = p1;
      l[rr] = l[rr] * rsc + rowsum16(p0 + p1);
      acc[0][rr] *= rsc; acc[1][rr] *= rsc; acc[2][rr] *= rsc; acc[3][rr] *= rsc;
    }

    // probs -> LDS (16x32 bf16), re-read as A fragment
    #pragma unroll
    for (int g = 0; g < 2; ++g) {
      #pragma unroll
      for (int rr = 0; rr < 8; ++rr) {
        pl[(rr + 8 * lh) * 32 + g * 16 + r] = (__bf16)s[g][rr];
      }
    }
    asm volatile("s_wait_dscnt 0" ::: "memory");
    v16bf ap = ld_frag(pl + r * 32 + bk);

    #pragma unroll
    for (int nt = 0; nt < 4; ++nt) {
      v16bf bv = ld_frag(Vh + (size_t)(nt * 16 + r) * VSTRIDE + j0 + bk);
      acc[nt] = wmma_bf16(ap, bv, acc[nt]);
    }
  }

  // finalize + store: path outputs occupy tokens [12000, 12281)
  #pragma unroll
  for (int rr = 0; rr < 8; ++rr) l[rr] = 1.0f / l[rr];
  int b = bh >> 3, h = bh & 7;
  float* ob = out + (size_t)b * NTOK * 512 + (size_t)h * 64;
  #pragma unroll
  for (int rr = 0; rr < 8; ++rr) {
    int prow = row0 + rr + 8 * lh;
    if (prow >= MPATH) continue;
    int tok = NHIST + prow;
    #pragma unroll
    for (int nt = 0; nt < 4; ++nt) {
      ob[(size_t)tok * 512 + nt * 16 + r] = acc[nt][rr] * l[rr];
    }
  }
}

// ------------------------------------------------------------------ launch ---

extern "C" void kernel_launch(void* const* d_in, const int* in_sizes, int n_in,
                              void* d_out, int out_size, void* d_ws, size_t ws_size,
                              hipStream_t stream) {
  (void)in_sizes; (void)n_in; (void)out_size; (void)ws_size;
  const float* x = (const float*)d_in[0];
  const float* W = (const float*)d_in[1];
  float* out = (float*)d_out;

  // Workspace layout (bf16). Region order matters: Kb before VTb so padded
  // column reads in K4 stay inside the workspace.
  char* ws = (char*)d_ws;
  size_t off = 0;
  __bf16* XB  = (__bf16*)(ws + off); off += (size_t)MROWS * DIM * 2;                    // 50.3 MB
  __bf16* WT  = (__bf16*)(ws + off); off += (size_t)NQKV * DIM * 2;                     //  1.5 MB
  __bf16* Qb  = (__bf16*)(ws + off); off += (size_t)BATCH * HEADS * NTOK * DHEAD * 2;   // 50.3 MB
  __bf16* Kb  = (__bf16*)(ws + off); off += (size_t)BATCH * HEADS * NTOK * DHEAD * 2;   // 50.3 MB
  __bf16* VTb = (__bf16*)(ws + off); off += (size_t)BATCH * HEADS * DHEAD * VSTRIDE * 2;// 50.4 MB

  // K1: conversions
  {
    int n = MROWS * DIM;
    cvt_x_kernel<<<(n + 255) / 256, 256, 0, stream>>>(x, XB, n);
    transpose_w_kernel<<<(NQKV * DIM + 255) / 256, 256, 0, stream>>>(W, WT);
  }
  // K2: QKV GEMM (32x64 tile per wave, double-buffered K loop)
  {
    int waves = ((MROWS + 31) / 32) * (NQKV / 64);  // 1536 * 24 = 36864
    qkv_gemm_kernel<<<(waves + 3) / 4, 128, 0, stream>>>(XB, WT, Qb, Kb, VTb);
  }
  // K3: hist-query attention over path keys/values -> tokens [0, 12000)
  {
    int waves = BATCH * HEADS * (NHIST / 16);       // 24000
    attn_hist_kernel<<<(waves + 3) / 4, 128, 0, stream>>>(Qb, Kb, VTb, out);
  }
  // K4: path-query attention over hist keys/values -> tokens [12000, 12281)
  {
    int waves = BATCH * HEADS * ((MPATH + 15) / 16);  // 576
    attn_path_kernel<<<(waves + 3) / 4, 128, 0, stream>>>(Qb, Kb, VTb, out);
  }
}
